// FDKN_84963043049985
// MI455X (gfx1250) — matrix operand order repaired and checked by reading, not compiled, not run on hardware
//
#include <hip/hip_runtime.h>
#include <cstdint>

typedef _Float16 f16;
typedef __attribute__((ext_vector_type(16))) _Float16 v16h;
typedef __attribute__((ext_vector_type(8)))  float    v8f;

// ---------------------------------------------------------------------------
// Repack: fold BN into conv weights/bias; emit f16 weights pre-swizzled into
// WMMA A-fragment order with K reordered as k = (dy*3+dx)*Cin + ci:
//   out[((t*steps + s)*32 + lane)*16 + j],  m = lane&15, h = lane>>4,
//   k = s*32 + 16h + j,  co = t*16 + m.
// Source (3x3): w[co*K + ci*9 + cell]  (OIHW => k_src = ci*9 + dy*3 + dx).
// Zero-padded for k >= K so the conv kernel needs no K guards on A.
// ---------------------------------------------------------------------------
__global__ __launch_bounds__(256) void repack_kernel(
    const float* __restrict__ w, const float* __restrict__ b,
    const float* __restrict__ g, const float* __restrict__ be,
    const float* __restrict__ m, const float* __restrict__ v,
    f16* __restrict__ wout, float* __restrict__ bout,
    int Cout, int K, int steps, int Cin, int c3)
{
  int idx = blockIdx.x * 256 + threadIdx.x;
  if (idx < Cout) {
    float a = 1.f, bb = b[idx];
    if (g) { a = g[idx] * rsqrtf(v[idx] + 1e-5f); bb = bb * a + be[idx] - m[idx] * a; }
    bout[idx] = bb;
  }
  int total = (Cout >> 4) * steps * 512;
  if (idx < total) {
    int j    = idx & 15;
    int ln   = (idx >> 4) & 31;
    int rest = idx >> 9;             // t*steps + s
    int s = rest % steps;
    int t = rest / steps;
    int hh = ln >> 4, mm = ln & 15;
    int k  = s * 32 + 16 * hh + j;
    int co = t * 16 + mm;
    float val = 0.f;
    if (k < K) {
      float a = 1.f;
      if (g) a = g[co] * rsqrtf(v[co] + 1e-5f);
      int src;
      if (c3) { int ci = k & (Cin - 1); int cell = k / Cin; src = co * K + ci * 9 + cell; }
      else    { src = co * K + k; }
      val = w[src] * a;
    }
    wout[idx] = (f16)val;
  }
}

// ---------------------------------------------------------------------------
// Prep: gray + space_to_depth(4) for both streams -> NHWC f16 [b][64*64][16]
// ---------------------------------------------------------------------------
__global__ __launch_bounds__(256) void prep_kernel(
    const float* __restrict__ image, const float* __restrict__ depth,
    f16* __restrict__ actIm, f16* __restrict__ actDp)
{
  int idx = blockIdx.x * 256 + threadIdx.x;
  if (idx >= 2 * 16 * 4096) return;
  int c = idx & 15;
  int p = (idx >> 4) & 4095;
  int b = idx >> 16;
  int ys = p >> 6, xs = p & 63;
  int yy = (ys << 2) + (c >> 2);
  int xx = (xs << 2) + (c & 3);
  int po = yy * 256 + xx;
  size_t ib = (size_t)b * 3 * 65536;
  float gray = 0.299f * image[ib + 2 * 65536 + po]
             + 0.587f * image[ib + 1 * 65536 + po]
             + 0.114f * image[ib + 0 * 65536 + po];
  size_t o = ((size_t)b * 4096 + p) * 16 + c;   // NHWC
  actIm[o] = (f16)gray;
  actDp[o] = (f16)depth[(size_t)b * 65536 + po];
}

// ---------------------------------------------------------------------------
// Implicit-GEMM conv via WMMA f32<-f16 16x16x32, NHWC activations, LDS-staged.
// Block = 256 threads = 8 waves: one output row y (4 N-tiles x 2 co-tiles).
// grid = (64 rows, ceil(Cout/32), batch).
// C3=true : 3x3 conv (rows y-1..y+1 staged, 1-pixel zero halo), f16+ReLU NHWC out.
// C3=false: 1x1 head (row y staged), f32 NHWC out, no ReLU.
// LDS layout: pixel-major, per-pixel channel stride = Cin+8 halves (16B pad
// breaks the 256B bank period => conflict-free b128 reads).
// B fragment = 16 consecutive channels of one (pixel,dy,dx) = 2x ds_load_b128.
// ---------------------------------------------------------------------------
template <bool C3>
__global__ __launch_bounds__(256) void conv_wmma_kernel(
    const f16* __restrict__ actIn, f16* __restrict__ outH, float* __restrict__ outF,
    const f16* __restrict__ Wfrag, const float* __restrict__ bias,
    int Cin, int Cout, int K, int steps, int lgC)
{
  // C3: 3*66*(128+8)=26928 halves; head: 64*(128+8)=8704 halves (in uint4)
  __shared__ uint4 sBuf[C3 ? 3366 : 1088];
  f16* sW = (f16*)sBuf;

  int tid  = threadIdx.x;
  int lane = tid & 31;
  int wv   = tid >> 5;
  int h = lane >> 4, l = lane & 15;
  int y   = blockIdx.x;              // 0..63
  int img = blockIdx.z;
  const f16* IN = actIn + (size_t)img * 4096 * Cin;   // NHWC

  const int stride = Cin + 8;        // halves per pixel in LDS
  const int vpp    = Cin >> 3;       // uint4 vectors per pixel
  const int lgv    = lgC - 3;

  // ---- cooperative LDS fill: global b128 -> ds_store_b128 ----
  if (C3) {
#pragma unroll
    for (int r = 0; r < 3; ++r) {
      int gy = y - 1 + r;
      const uint4* src = (const uint4*)(IN + (size_t)(gy << 6) * Cin);
      int rbase = r * 66;
      for (int i = tid; i < (vpp << 6); i += 256) {
        int px  = i >> lgv;
        int ci8 = i & (vpp - 1);
        uint4 val = {0u, 0u, 0u, 0u};
        if ((unsigned)gy < 64u) val = src[i];
        *(uint4*)(sW + (size_t)(rbase + 1 + px) * stride + (ci8 << 3)) = val;
      }
      for (int i = tid; i < vpp; i += 256) {        // zero halo pixels
        uint4 z = {0u, 0u, 0u, 0u};
        *(uint4*)(sW + (size_t)rbase * stride + (i << 3)) = z;
        *(uint4*)(sW + (size_t)(rbase + 65) * stride + (i << 3)) = z;
      }
    }
  } else {
    const uint4* src = (const uint4*)(IN + (size_t)(y << 6) * Cin);
    for (int i = tid; i < (vpp << 6); i += 256) {
      int px  = i >> lgv;
      int ci8 = i & (vpp - 1);
      *(uint4*)(sW + (size_t)px * stride + (ci8 << 3)) = src[i];
    }
  }
  __syncthreads();

  int nt = wv & 3;                     // N-tile within row
  int t  = blockIdx.y * 2 + (wv >> 2); // co-tile
  if (t * 16 >= Cout) return;          // pad waves (heads: Cout=144 -> 9 tiles)
  int xN  = (nt << 4) + l;             // this lane's pixel column (N index)
  int co0 = t << 4;

  const v16h* Afrag = (const v16h*)Wfrag + (size_t)t * steps * 32 + lane;

  v8f acc = {};
  for (int s = 0; s < steps; ++s) {
    __builtin_prefetch(Afrag + (s + 1) * 32, 0, 0);   // global_prefetch_b8
    v16h av = Afrag[s * 32];           // A-fragment: one contiguous 32B load
    // B-fragment: 16 consecutive channels (K-order = cell*Cin + ci)
    int kk = s * 32 + 16 * h;          // this half-wave's K base
    v16h bv = {};
    if (kk < K) {                      // only L1 (K=144) has a padded tail
      int base;
      if (C3) {
        int cell = kk >> lgC;
        int cio  = kk & (Cin - 1);
        int dy = cell / 3;
        int dx = cell - dy * 3;
        base = (dy * 66 + 1 + (xN + dx - 1)) * stride + cio;
      } else {
        base = xN * stride + kk;
      }
      union { uint4 q[2]; v16h v; } ub;
      const uint4* p = (const uint4*)(sW + base);     // 16B aligned
      ub.q[0] = p[0];
      ub.q[1] = p[1];
      bv = ub.v;
    }
    acc = __builtin_amdgcn_wmma_f32_16x16x32_f16(
        /*neg_a=*/false, av, /*neg_b=*/false, bv,
        /*c_mod=*/(short)0, acc, /*reuse_a=*/false, /*reuse_b=*/false);
  }

  // D layout: lane 16h+l, VGPR r -> (M = 8h + r, N = l). NHWC stores:
  // 8 consecutive channels per pixel => single b128 (f16) / 2x float4 (f32).
  int pIdx = (y << 6) + xN;
  if (C3) {
    union { uint4 q; f16 hx[8]; } pk;
#pragma unroll
    for (int r = 0; r < 8; ++r) {
      float o = acc[r] + bias[co0 + (h << 3) + r];
      pk.hx[r] = (f16)fmaxf(o, 0.f);
    }
    *(uint4*)(outH + ((size_t)img * 4096 + pIdx) * Cout + co0 + (h << 3)) = pk.q;
  } else {
    float* dst = outF + ((size_t)img * 4096 + pIdx) * Cout + co0 + (h << 3);
    float4 f0, f1;
    f0.x = acc[0] + bias[co0 + (h << 3) + 0];
    f0.y = acc[1] + bias[co0 + (h << 3) + 1];
    f0.z = acc[2] + bias[co0 + (h << 3) + 2];
    f0.w = acc[3] + bias[co0 + (h << 3) + 3];
    f1.x = acc[4] + bias[co0 + (h << 3) + 4];
    f1.y = acc[5] + bias[co0 + (h << 3) + 5];
    f1.z = acc[6] + bias[co0 + (h << 3) + 6];
    f1.w = acc[7] + bias[co0 + (h << 3) + 7];
    *(float4*)dst       = f0;
    *((float4*)dst + 1) = f1;
  }
}

// ---------------------------------------------------------------------------
// Final: sigmoid/product heads (NHWC), pixel-shuffle gather, mean-subtract,
// deformable bilinear sampling from 15x15 depth patches, residual add.
// ---------------------------------------------------------------------------
__device__ __forceinline__ float tap15(const float* __restrict__ D,
                                       int y, int x, int py, int px)
{
  if ((unsigned)py > 14u || (unsigned)px > 14u) return 0.f;     // patch bounds
  int row = y - 7 + py, col = x - 7 + px;
  if ((unsigned)row > 255u || (unsigned)col > 255u) return 0.f; // image zero-pad
  return D[row * 256 + col];
}

__global__ __launch_bounds__(256) void final_kernel(
    const float* __restrict__ depth,
    const float* __restrict__ hwIm, const float* __restrict__ hoIm,
    const float* __restrict__ hwDp, const float* __restrict__ hoDp,
    float* __restrict__ out)
{
  int idx = blockIdx.x * 256 + threadIdx.x;
  if (idx >= 2 * 256 * 256) return;
  int x = idx & 255, y = (idx >> 8) & 255, b = idx >> 16;
  int ys = y >> 2, xs = x >> 2;
  int sp = ((y & 3) << 2) | (x & 3);      // pixel-shuffle sub-channel
  int pix = (ys << 6) | xs;

  const float* HWi = hwIm + ((size_t)b * 4096 + pix) * 144;  // NHWC rows
  const float* HWd = hwDp + ((size_t)b * 4096 + pix) * 144;
  const float* HOi = hoIm + ((size_t)b * 4096 + pix) * 288;
  const float* HOd = hoDp + ((size_t)b * 4096 + pix) * 288;

  float wgt[9], wsum = 0.f;
#pragma unroll
  for (int c = 0; c < 9; ++c) {
    int u = c * 16 + sp;
    float a = HWi[u];
    float d = HWd[u];
    float val = (1.f / (1.f + __expf(-a))) * (1.f / (1.f + __expf(-d)));
    wgt[c] = val; wsum += val;
  }
  float wmean = wsum * (1.f / 9.f);

  const float* D = depth + (size_t)b * 65536;
  float acc = 0.f;
#pragma unroll
  for (int ri = 0; ri < 3; ++ri) {
#pragma unroll
    for (int rj = 0; rj < 3; ++rj) {
      int cc = ri * 3 + rj;
      int u0 = (cc * 2 + 0) * 16 + sp;
      int u1 = (cc * 2 + 1) * 16 + sp;
      float o0 = HOi[u0] * HOd[u0];
      float o1 = HOi[u1] * HOd[u1];
      // ix(col) = 7 + ri + o0 - 0.5 ; iy(row) = 7 + rj + o1 - 0.5
      float ixf = 6.5f + (float)ri + o0;
      float iyf = 6.5f + (float)rj + o1;
      float xf0 = floorf(ixf), yf0 = floorf(iyf);
      int px0 = (int)xf0, py0 = (int)yf0;
      float wx1 = ixf - xf0, wx0 = 1.f - wx1;
      float wy1 = iyf - yf0, wy0 = 1.f - wy1;
      float s00 = tap15(D, y, x, py0,     px0);
      float s01 = tap15(D, y, x, py0,     px0 + 1);
      float s10 = tap15(D, y, x, py0 + 1, px0);
      float s11 = tap15(D, y, x, py0 + 1, px0 + 1);
      float samp = s00 * wy0 * wx0 + s01 * wy0 * wx1 + s10 * wy1 * wx0 + s11 * wy1 * wx1;
      acc += (wgt[cc] - wmean) * samp;
    }
  }
  out[idx] = acc + D[y * 256 + x];
}

// ---------------------------------------------------------------------------
// Host orchestration.
// d_in: [0]=image (2,3,256,256) f32, [1]=depth (2,1,256,256) f32,
//       [2..29]=im_params (w1,b1,g1,be1,m1,v1,w2,b2,w3,b3,g3..v3,w4,b4,
//                          w5,b5,g5..v5,w6,b6,ww,bw,wo,bo), [30..57]=dp_params.
// ---------------------------------------------------------------------------
extern "C" void kernel_launch(void* const* d_in, const int* in_sizes, int n_in,
                              void* d_out, int out_size, void* d_ws, size_t ws_size,
                              hipStream_t stream)
{
  (void)in_sizes; (void)n_in; (void)out_size; (void)ws_size;
  const float* image = (const float*)d_in[0];
  const float* depth = (const float*)d_in[1];

  size_t off = 0;
  auto alloc = [&](size_t bytes) -> void* {
    off = (off + 255) & ~(size_t)255;
    void* p = (char*)d_ws + off;
    off += bytes;
    return p;
  };

  f16*   wpk[2];
  float* bfd[2];
  for (int s = 0; s < 2; ++s) {
    wpk[s] = (f16*)alloc((size_t)346112 * sizeof(f16)); // fragment-packed, per stream
    bfd[s] = (float*)alloc(880 * sizeof(float));        // folded biases, per stream
  }
  f16* prepA[2];
  prepA[0] = (f16*)alloc((size_t)2 * 4096 * 16 * sizeof(f16));
  prepA[1] = (f16*)alloc((size_t)2 * 4096 * 16 * sizeof(f16));
  f16* actA = (f16*)alloc((size_t)2 * 4096 * 128 * sizeof(f16));
  f16* actB = (f16*)alloc((size_t)2 * 4096 * 128 * sizeof(f16));
  float* headW[2];
  float* headO[2];
  for (int s = 0; s < 2; ++s) {
    headW[s] = (float*)alloc((size_t)2 * 4096 * 144 * sizeof(float));
    headO[s] = (float*)alloc((size_t)2 * 4096 * 288 * sizeof(float));
  }

  struct LD { int w, b, g, Cin, Cout, K, steps, lgC, wOff, bOff; };
  static const LD L[8] = {
    { 0,  1,  2,  16,  32,  144,  5, 4,      0,   0},  // conv1 + bn1
    { 6,  7, -1,  32,  32,  288,  9, 5,   5120,  32},  // conv2
    { 8,  9, 10,  32,  64,  288,  9, 5,  14336,  64},  // conv3 + bn3
    {14, 15, -1,  64,  64,  576, 18, 6,  32768, 128},  // conv4
    {16, 17, 18,  64, 128,  576, 18, 6,  69632, 192},  // conv5 + bn5
    {22, 23, -1, 128, 128, 1152, 36, 7, 143360, 320},  // conv6
    {24, 25, -1, 128, 144,  128,  4, 7, 290816, 448},  // weight head (1x1)
    {26, 27, -1, 128, 288,  128,  4, 7, 309248, 592},  // offset head (1x1)
  };

  // 1) fold BN + pack weights into WMMA fragment order (reordered K)
  for (int s = 0; s < 2; ++s) {
    int base = 2 + s * 28;
    for (int li = 0; li < 8; ++li) {
      const LD& d = L[li];
      const float *g = nullptr, *be = nullptr, *m = nullptr, *vv = nullptr;
      if (d.g >= 0) {
        g  = (const float*)d_in[base + d.g];
        be = (const float*)d_in[base + d.g + 1];
        m  = (const float*)d_in[base + d.g + 2];
        vv = (const float*)d_in[base + d.g + 3];
      }
      int total = (d.Cout >> 4) * d.steps * 512;
      repack_kernel<<<(total + 255) / 256, 256, 0, stream>>>(
          (const float*)d_in[base + d.w], (const float*)d_in[base + d.b],
          g, be, m, vv, wpk[s] + d.wOff, bfd[s] + d.bOff,
          d.Cout, d.K, d.steps, d.Cin, (li < 6) ? 1 : 0);
    }
  }

  // 2) gray + space_to_depth for both streams (NHWC)
  prep_kernel<<<(2 * 16 * 4096 + 255) / 256, 256, 0, stream>>>(
      image, depth, prepA[0], prepA[1]);

  // 3) kernel-net per stream (im=0, dp=1), WMMA implicit GEMM convs
  for (int s = 0; s < 2; ++s) {
    const f16* cur = prepA[s];
    f16* bufs[2] = {actA, actB};
    for (int li = 0; li < 6; ++li) {
      const LD& d = L[li];
      f16* outp = bufs[li & 1];
      int coGroups = (d.Cout / 16 + 1) / 2;
      conv_wmma_kernel<true><<<dim3(64, coGroups, 2), 256, 0, stream>>>(
          cur, outp, nullptr, wpk[s] + d.wOff, bfd[s] + d.bOff,
          d.Cin, d.Cout, d.K, d.steps, d.lgC);
      cur = outp;
    }
    conv_wmma_kernel<false><<<dim3(64, 5, 2), 256, 0, stream>>>(   // 9 tiles
        cur, nullptr, headW[s], wpk[s] + L[6].wOff, bfd[s] + L[6].bOff,
        128, 144, 128, 4, 7);
    conv_wmma_kernel<false><<<dim3(64, 9, 2), 256, 0, stream>>>(   // 18 tiles
        cur, nullptr, headO[s], wpk[s] + L[7].wOff, bfd[s] + L[7].bOff,
        128, 288, 128, 4, 7);
  }

  // 4) fuse heads + deformable sampling + residual
  final_kernel<<<(2 * 256 * 256 + 255) / 256, 256, 0, stream>>>(
      depth, headW[0], headO[0], headW[1], headO[1], (float*)d_out);
}
